// SEMConv_86973087744453
// MI455X (gfx1250) — compile-verified, matrix operand order checked by reading, not run on hardware
//
#include <hip/hip_runtime.h>

typedef __attribute__((ext_vector_type(16))) _Float16 v16h;
typedef __attribute__((ext_vector_type(8)))  _Float16 v8h;
typedef __attribute__((ext_vector_type(4)))  _Float16 v4h;
typedef __attribute__((ext_vector_type(8)))  float    v8f;

#define NE      144   // elements per dim
#define MPTS    8     // nodes per element dim (m)
#define S1      5     // window taps
#define RAD     2
#define MH      33    // n//2 + 1 for n=64 modes
#define KPITCH  136   // 128 + 8 f16 pad -> 272B row pitch (conflict-free b128)
#define SLABROWS 160  // (16+4) halo-dim x 8 other-dim
// gpack: [dir][XY 8][kk 5][kb 4][lane 32][16 f16]  = 327,680 B in d_ws
#define GP_FRAG_F16 512  // 32 lanes * 16 f16 per fragment

// ---------------------------------------------------------------------------
// Phase 1: build G[k,X,x,i,o] = fourier_kernel(coef, xarg)[i,o] * ws[k,X,x],
// converted to f16 and pre-swizzled into per-lane WMMA B-fragment layout.
// ---------------------------------------------------------------------------
__global__ void __launch_bounds__(256)
prep_g(const float* __restrict__ coef0, const float* __restrict__ coef1,
       const float* __restrict__ ws0,   const float* __restrict__ ws1,
       const float* __restrict__ grid,  _Float16* __restrict__ gpack)
{
    __shared__ float cs[MPTS][MH];
    __shared__ float sn[MPTS][MH];

    const int blk = blockIdx.x;          // 0..79 = dir(2) * X(8) * kk(5)
    const int dir = blk / 40;
    const int rem = blk - dir * 40;
    const int X   = rem / S1;
    const int kk  = rem - X * S1;
    const float* __restrict__ coef = dir ? coef1 : coef0;
    const float* __restrict__ ws   = dir ? ws1  : ws0;
    const int tid = threadIdx.x;

    const float gX = grid[X];
    for (int idx = tid; idx < MPTS * MH; idx += 256) {
        int x  = idx / MH;
        int mm = idx - x * MH;
        float xa = (gX - grid[x] - (float)(kk - RAD)) * 0.25f;
        float th = 6.28318530717958647692f * (float)mm * xa;
        float s, c;
        __sincosf(th, &s, &c);
        float scale = (mm == 0 || mm == MH - 1) ? 1.0f : 2.0f;
        cs[x][mm] = c * scale;
        sn[x][mm] = s * scale;
    }
    __syncthreads();

    for (int idx = tid; idx < 2048; idx += 256) {   // 4 kb * 32 lanes * 16 elems
        int kb = idx >> 9;
        int l  = (idx >> 4) & 31;
        int j  = idx & 15;
        int hi8 = (l & 16) ? 8 : 0;                 // lanes 16-31 hold K+8 half
        int K = kb * 32 + ((j < 8) ? j : j + 8) + hi8;
        int x = K >> 4;                             // reduction outer index
        int i = K & 15;                             // input channel
        int o = l & 15;                             // output channel (N)
        const float* __restrict__ cio = coef + i * 16 + o;

        float acc = cs[x][0] * cio[0];              // m=0: ii=0, scale=1
        #pragma unroll 4
        for (int mm = 1; mm <= 31; ++mm)            // interior: rr - ii terms
            acc += cs[x][mm] * cio[mm * 256] - sn[x][mm] * cio[(64 - mm) * 256];
        acc += cs[x][32] * cio[32 * 256];           // m=32: ii=0, scale=1

        float wsv = ws[(kk * MPTS + X) * MPTS + x];
        size_t off = ((((size_t)dir * 8 + X) * S1 + kk) * 4 + kb) * GP_FRAG_F16
                   + (size_t)l * 16 + j;
        gpack[off] = (_Float16)(acc * wsv);
    }
}

// ---------------------------------------------------------------------------
// Phase 2: convolution-GEMM.  DIR=0: rows (y,a,b), halo along a, cols (X,o).
//                             DIR=1: rows (x,a,b), halo along b, cols (Y,o).
// Block = 256 threads = 8 waves; wave w handles column block X(Y)=w.
// kk loop rolled (5 iters, ~160 VGPRs); per kk: 4 B fragments batched (one
// B-wait per 32 WMMAs) + next-kk B prefetch; A fragments distance-1
// double-buffered so ds_load(next) overlaps wmma(cur).
// ---------------------------------------------------------------------------
template <int DIR>
__global__ void __launch_bounds__(256)
semconv_kernel(const float* __restrict__ nodal,
               const _Float16* __restrict__ gpack,
               float* __restrict__ out)
{
    __shared__ __align__(16) _Float16 lds[SLABROWS * KPITCH];

    const int tid = threadIdx.x;
    const int bz  = blockIdx.z;                    // y (DIR0) / x (DIR1)
    int a0, b0;
    if constexpr (DIR == 0) { a0 = blockIdx.x * 16; b0 = blockIdx.y * 8;  }
    else                    { a0 = blockIdx.x * 8;  b0 = blockIdx.y * 16; }

    // ---- stage 1: load slab (fp32 -> f16) into LDS, K=(outer*16+i) packed
    for (int t = tid; t < 5120; t += 256) {        // 8 outer * 20 halo * 8 * 4
        int i4 = t & 3;
        size_t gidx;
        int lrow, koff;
        if constexpr (DIR == 0) {
            int b  = (t >> 2) & 7;
            int ap = (t >> 5) % 20;
            int x  = t / 640;
            int ga = (a0 + ap - RAD + NE) % NE;    // circular halo in a
            gidx = ((((size_t)x * MPTS + bz) * NE + ga) * NE + (b0 + b)) * 16 + i4 * 4;
            lrow = ap * 8 + b;
            koff = x * 16 + i4 * 4;
        } else {
            int bp = (t >> 2) % 20;
            int a  = (t / 80) & 7;
            int yv = t / 640;
            int gb = (b0 + bp - RAD + NE) % NE;    // circular halo in b
            gidx = ((((size_t)bz * MPTS + yv) * NE + (a0 + a)) * NE + gb) * 16 + i4 * 4;
            lrow = bp * 8 + a;
            koff = yv * 16 + i4 * 4;
        }
        const float4 v = *(const float4*)(nodal + gidx);
        v4h h = { (_Float16)v.x, (_Float16)v.y, (_Float16)v.z, (_Float16)v.w };
        *(v4h*)&lds[lrow * KPITCH + koff] = h;
    }
    __syncthreads();

    // ---- stage 2: WMMA core
    const int wave  = tid >> 5;                    // = X (DIR0) / Y (DIR1)
    const int lane  = tid & 31;
    const int mlane = lane & 15;
    const int hi8   = (lane & 16) ? 8 : 0;

    v8f acc[8];
    const v8f zero = {0.f, 0.f, 0.f, 0.f, 0.f, 0.f, 0.f, 0.f};
    #pragma unroll
    for (int q = 0; q < 8; ++q) acc[q] = zero;

    const v16h* __restrict__ gf =
        (const v16h*)gpack + ((size_t)DIR * 8 + wave) * (S1 * 4 * 32) + lane;

    v16h B[4], Bn[4];
    #pragma unroll
    for (int q = 0; q < 4; ++q) B[q] = gf[q * 32];

    #pragma unroll 1
    for (int kk = 0; kk < S1; ++kk) {
        // prefetch next kk's 4 B fragments (clamped on last iteration)
        const int kkn = (kk < S1 - 1) ? kk + 1 : kk;
        #pragma unroll
        for (int q = 0; q < 4; ++q) Bn[q] = gf[(kkn * 4 + q) * 32];

        // window shift folds linearly: slab row = 16*mt + 8*kk + m
        const _Float16* __restrict__ kbase =
            &lds[(kk * 8 + mlane) * KPITCH + hi8];

        v8h lo = *(const v8h*)kbase;               // A(kb=0,mt=0)
        v8h hi = *(const v8h*)(kbase + 16);
        #pragma unroll
        for (int s = 0; s < 32; ++s) {             // kb = s>>3, mt = s&7
            const int kb = s >> 3, mt = s & 7;
            const int sn  = (s < 31) ? s + 1 : s;
            const int kbn = sn >> 3, mtn = sn & 7;
            const _Float16* apn = kbase + kbn * 32 + mtn * 16 * KPITCH;
            v8h lo2 = *(const v8h*)apn;            // ds_load_b128 (next)
            v8h hi2 = *(const v8h*)(apn + 16);     // ds_load_b128 (next)
            v16h A = __builtin_shufflevector(
                lo, hi, 0,1,2,3,4,5,6,7,8,9,10,11,12,13,14,15);
            acc[mt] = __builtin_amdgcn_wmma_f32_16x16x32_f16(
                false, A, false, B[kb], (short)0, acc[mt], false, false);
            lo = lo2; hi = hi2;
        }
        #pragma unroll
        for (int q = 0; q < 4; ++q) B[q] = Bn[q];
    }

    // ---- stage 3: writeback (DIR0 stores, DIR1 read-adds; stream-ordered)
    // mrow = 16*mt + r + 8*h  (h = lane>=16)  ->  a_l/b_l = 2*mt + h, r
    const int o = lane & 15;
    const int h = (lane >> 4) & 1;
    if constexpr (DIR == 0) {
        float* __restrict__ po =
            out + ((((size_t)wave * MPTS + bz) * NE + (a0 + h)) * NE + b0) * 16 + o;
        #pragma unroll
        for (int mt = 0; mt < 8; ++mt) {
            #pragma unroll
            for (int r = 0; r < 8; ++r)
                po[mt * 2 * NE * 16 + r * 16] = acc[mt][r];
        }
    } else {
        float* __restrict__ po =
            out + ((((size_t)bz * MPTS + wave) * NE + a0) * NE + (b0 + h)) * 16 + o;
        #pragma unroll
        for (int mt = 0; mt < 8; ++mt) {
            #pragma unroll
            for (int r = 0; r < 8; ++r)
                po[r * NE * 16 + mt * 2 * 16] += acc[mt][r];
        }
    }
}

// ---------------------------------------------------------------------------
extern "C" void kernel_launch(void* const* d_in, const int* in_sizes, int n_in,
                              void* d_out, int out_size, void* d_ws, size_t ws_size,
                              hipStream_t stream)
{
    (void)in_sizes; (void)n_in; (void)out_size; (void)ws_size;
    const float* nodal = (const float*)d_in[0];
    const float* coef0 = (const float*)d_in[1];
    const float* coef1 = (const float*)d_in[2];
    const float* ws0   = (const float*)d_in[3];
    const float* ws1   = (const float*)d_in[4];
    const float* grid  = (const float*)d_in[5];
    float*       out   = (float*)d_out;
    _Float16*    gpack = (_Float16*)d_ws;          // 327,680 B used

    prep_g<<<80, 256, 0, stream>>>(coef0, coef1, ws0, ws1, grid, gpack);
    // dir0: a-tiles(144/16=9) x b-tiles(144/8=18) x y(8)
    semconv_kernel<0><<<dim3(9, 18, 8), 256, 0, stream>>>(nodal, gpack, out);
    // dir1: a-tiles(18) x b-tiles(9) x x(8), accumulates into out
    semconv_kernel<1><<<dim3(18, 9, 8), 256, 0, stream>>>(nodal, gpack, out);
}
